// SAE_7859790152355
// MI455X (gfx1250) — matrix operand order repaired and checked by reading, not compiled, not run on hardware
//
#include <hip/hip_runtime.h>
#include <hip/hip_bf16.h>

// ---------------------------------------------------------------------------
// TopK-SAE forward for MI455X (gfx1250, wave32, WMMA, async-to-LDS).
//   z = (x - bd) @ Ae^T          bf16 WMMA, f32 accum, LDS double-buffered
//                                via global_load_async_to_lds_b128 (ASYNCcnt),
//                                non-temporal z stores (streaming, > L2)
//   topk(relu(z), 64) per row    radix-select, row staged once into LDS
//   out = lam * codes @ Ad^T + bd  sparse gather over pre-transposed Ad
// ---------------------------------------------------------------------------

#define B_ROWS 8192
#define DIN    1024
#define WIDTH  16384
#define TOPK   64

typedef __attribute__((ext_vector_type(16))) __bf16 v16bf;
typedef __attribute__((ext_vector_type(8)))  float  v8f;

union FragAB {
    v16bf bf;
    uint4 u4[2];
};

// workspace layout (bytes)
#define XCB_OFF 0ull                       // 8192*1024*2  = 16 MiB  (bf16 x-bd)
#define AEB_OFF 16777216ull                // 16384*1024*2 = 32 MiB  (bf16 Ae)
#define ADT_OFF 50331648ull                // 16384*1024*4 = 64 MiB  (Ad transposed)
#define Z_OFF   117440512ull               // 8192*16384*4 = 512 MiB (pre-codes)
#define IDX_OFF 654311424ull               // 8192*64*4    = 2 MiB
#define VAL_OFF 656408576ull               // 8192*64*4    = 2 MiB

__device__ __forceinline__ unsigned short f32_to_bf16_rne(float f) {
    unsigned int u = __float_as_uint(f);
    unsigned int r = 0x7FFFu + ((u >> 16) & 1u);
    return (unsigned short)((u + r) >> 16);
}

// LDS byte offset of a __shared__ pointer: generic LDS addresses keep the
// wave-relative offset in addr[31:0] (CDNA5 flat-addressing aperture rules).
__device__ __forceinline__ unsigned int lds_offset(const void* p) {
    return (unsigned int)(unsigned long long)(size_t)p;
}

// async copy: 16 bytes/lane global -> LDS, tracked by ASYNCcnt
__device__ __forceinline__ void async_copy_b128(unsigned int lds_off,
                                                const void* gaddr) {
    asm volatile("global_load_async_to_lds_b128 %0, %1, off"
                 :
                 : "v"(lds_off), "v"((unsigned long long)(size_t)gaddr)
                 : "memory");
}

// ---------------- prep: xcb = bf16(x - bd) --------------------------------
__global__ void sae_prep_x(const float* __restrict__ x,
                           const float* __restrict__ bd,
                           unsigned short* __restrict__ xcb) {
    int i = blockIdx.x * blockDim.x + threadIdx.x;      // < 8192*1024
    int d = i & (DIN - 1);
    xcb[i] = f32_to_bf16_rne(x[i] - bd[d]);
}

// ---------------- prep: aeb = bf16(Ae) ------------------------------------
__global__ void sae_prep_ae(const float* __restrict__ ae,
                            unsigned short* __restrict__ aeb) {
    int i = blockIdx.x * blockDim.x + threadIdx.x;      // < 16384*1024
    aeb[i] = f32_to_bf16_rne(ae[i]);
}

// ---------------- prep: Adt[j][d] = Ad[d][j] ------------------------------
__global__ void sae_trans_ad(const float* __restrict__ ad,   // (1024, 16384)
                             float* __restrict__ adt) {      // (16384, 1024)
    __shared__ float tile[32][33];
    int tx = threadIdx.x, ty = threadIdx.y;                  // 32 x 8
    int j0 = blockIdx.x * 32;                                // WIDTH dim
    int d0 = blockIdx.y * 32;                                // DIN dim
#pragma unroll
    for (int q = 0; q < 4; ++q) {
        int d = d0 + ty + q * 8;
        tile[ty + q * 8][tx] = ad[d * WIDTH + (j0 + tx)];
    }
    __syncthreads();
#pragma unroll
    for (int q = 0; q < 4; ++q) {
        int j = j0 + ty + q * 8;
        adt[j * DIN + (d0 + tx)] = tile[tx][ty + q * 8];
    }
}

// ---------------- encode GEMM: z = xcb @ aeb^T (bf16 WMMA) ----------------
// block = 256 threads = 8 waves (4 in M x 2 in N). WG tile 128M x 128N.
// K stepped by 32; A/B tiles double-buffered in LDS via async-to-LDS with a
// statically unrolled 2-stage pipeline (no dynamic buffer index, no acc phis).
// LDS rows padded to 80 B (16B-aligned b128 reads, de-phased banks).
#define ENC_ROWPITCH 80
#define ENC_TILEBYTES (128 * ENC_ROWPITCH)     // 10240 B per tile

__global__ __launch_bounds__(256)
void sae_encode(const unsigned short* __restrict__ xcb,   // (8192,1024) bf16
                const unsigned short* __restrict__ aeb,   // (16384,1024) bf16
                float* __restrict__ z) {                  // (8192,16384) f32
    __shared__ __align__(16) unsigned char smem[2][2][ENC_TILEBYTES]; // 40 KB

    const int tid  = threadIdx.x;
    const int wave = tid >> 5;
    const int lane = tid & 31;
    const int wm   = wave & 3;          // 0..3 (M)
    const int wn   = wave >> 2;         // 0..1 (N)
    const int mb   = blockIdx.y * 128;
    const int nb   = blockIdx.x * 128;
    const int l    = lane & 15;
    const int hi   = lane >> 4;

    // This thread's two 16B chunks of each 128x32 bf16 tile (512 chunks total)
    // chunk q: row = q>>2, 16B-column c = q&3.
    auto issue_stage = [&](int s, int k0) {
#pragma unroll
        for (int j = 0; j < 2; ++j) {
            int q   = tid * 2 + j;
            int row = q >> 2;
            int c   = q & 3;
            async_copy_b128(
                lds_offset(&smem[s][0][row * ENC_ROWPITCH + c * 16]),
                xcb + (size_t)(mb + row) * DIN + k0 + c * 8);
            async_copy_b128(
                lds_offset(&smem[s][1][row * ENC_ROWPITCH + c * 16]),
                aeb + (size_t)(nb + row) * DIN + k0 + c * 8);
        }
    };

    v8f zero = {};
    v8f acc[2][4];
#pragma unroll
    for (int mi = 0; mi < 2; ++mi)
#pragma unroll
        for (int ni = 0; ni < 4; ++ni) acc[mi][ni] = zero;

    // 8 WMMAs against buffer s
    auto compute_stage = [&](int s) {
        const unsigned char* As = smem[s][0];
        const unsigned char* Bs = smem[s][1];
        // A frag: lane l holds row (wm*32+mi*16+l); K hi*8..+7 then +16
        FragAB fa[2];
#pragma unroll
        for (int mi = 0; mi < 2; ++mi) {
            const unsigned char* pa =
                As + (wm * 32 + mi * 16 + l) * ENC_ROWPITCH + hi * 16;
            fa[mi].u4[0] = *(const uint4*)(pa);
            fa[mi].u4[1] = *(const uint4*)(pa + 32);
        }
        // B frag: lane l holds col (wn*64+ni*16+l); K hi*8..+7 then +16
        FragAB fb[4];
#pragma unroll
        for (int ni = 0; ni < 4; ++ni) {
            const unsigned char* pb =
                Bs + (wn * 64 + ni * 16 + l) * ENC_ROWPITCH + hi * 16;
            fb[ni].u4[0] = *(const uint4*)(pb);
            fb[ni].u4[1] = *(const uint4*)(pb + 32);
        }
#pragma unroll
        for (int mi = 0; mi < 2; ++mi)
#pragma unroll
            for (int ni = 0; ni < 4; ++ni)
                acc[mi][ni] = __builtin_amdgcn_wmma_f32_16x16x32_bf16(
                    false, fa[mi].bf, false, fb[ni].bf,
                    (short)0, acc[mi][ni], false, false);
    };

    const int NSTEP = DIN / 32;                 // 32 K-steps (even)
    issue_stage(0, 0);

    // steady state: two stages per trip, static buffer ids
#pragma unroll 1
    for (int ks = 0; ks < NSTEP - 2; ks += 2) {
        issue_stage(1, (ks + 1) * 32);
        asm volatile("s_wait_asynccnt 0x4" ::: "memory");  // buf0 landed
        __syncthreads();
        compute_stage(0);
        __syncthreads();                                   // buf0 free

        issue_stage(0, (ks + 2) * 32);
        asm volatile("s_wait_asynccnt 0x4" ::: "memory");  // buf1 landed
        __syncthreads();
        compute_stage(1);
        __syncthreads();                                   // buf1 free
    }
    // drain: stages NSTEP-2 (buf0 in flight) and NSTEP-1
    issue_stage(1, (NSTEP - 1) * 32);
    asm volatile("s_wait_asynccnt 0x4" ::: "memory");
    __syncthreads();
    compute_stage(0);
    __syncthreads();
    asm volatile("s_wait_asynccnt 0x0" ::: "memory");
    __syncthreads();
    compute_stage(1);

    // C/D layout: VGPR r, lanes 0-15 -> (M=r, N=lane); lanes 16-31 -> (M=r+8).
    // z is streaming (512 MB, written once, read once much later): store NT
    // so xcb/aeb stay resident in L2 for the remaining workgroups.
#pragma unroll
    for (int mi = 0; mi < 2; ++mi)
#pragma unroll
        for (int ni = 0; ni < 4; ++ni) {
            int row_base = mb + wm * 32 + mi * 16 + hi * 8;
            int col      = nb + wn * 64 + ni * 16 + l;
#pragma unroll
            for (int r = 0; r < 8; ++r)
                __builtin_nontemporal_store(
                    acc[mi][ni][r],
                    &z[(size_t)(row_base + r) * WIDTH + col]);
        }
}

// ---------------- per-row top-64 radix select + compaction ----------------
// Row (64 KB) staged once into LDS via async-to-LDS; all 6 passes hit LDS.
__global__ __launch_bounds__(256)
void sae_topk(const float* __restrict__ z,
              int* __restrict__ idx_out,          // (8192, 64)
              float* __restrict__ val_out) {      // (8192, 64)
    __shared__ __align__(16) float zrow[WIDTH];   // 64 KB
    __shared__ unsigned int hist[256];
    __shared__ unsigned int sh_sel, sh_want, sh_cnt;

    const int row = blockIdx.x;
    const int tid = threadIdx.x;
    const float* zr = z + (size_t)row * WIDTH;
    int*   irow = idx_out + row * TOPK;
    float* vrow = val_out + row * TOPK;

    // stage the row: 4096 x 16B chunks, 16 per thread
#pragma unroll
    for (int j = 0; j < 16; ++j) {
        int c = tid + j * 256;
        async_copy_b128(lds_offset(&zrow[c * 4]), zr + c * 4);
    }
    asm volatile("s_wait_asynccnt 0x0" ::: "memory");

    // default-fill (rows with <64 survivors still produce defined output)
    if (tid < TOPK) { irow[tid] = 0; vrow[tid] = 0.0f; }
    __syncthreads();

    unsigned int prefix = 0;
    unsigned int want   = TOPK;
    for (int shift = 24; shift >= 0; shift -= 8) {
        hist[tid] = 0;
        __syncthreads();
        unsigned int maskhi = (shift == 24) ? 0u : (0xFFFFFFFFu << (shift + 8));
        for (int i = tid; i < WIDTH; i += 256) {
            float zv = zrow[i];
            unsigned int key = __float_as_uint(zv > 0.0f ? zv : 0.0f);
            if ((key & maskhi) == prefix)
                atomicAdd(&hist[(key >> shift) & 255], 1u);
        }
        __syncthreads();
        if (tid == 0) {
            unsigned int cum = 0, sel = 0, w = want;
            for (int b = 255; b >= 0; --b) {
                unsigned int c = hist[b];
                if (cum + c >= w) { sel = (unsigned int)b; w -= cum; break; }
                cum += c;
            }
            sh_sel = sel; sh_want = w;
        }
        __syncthreads();
        prefix |= (sh_sel << shift);
        want = sh_want;
        __syncthreads();
    }
    const unsigned int thresh = prefix;

    if (tid == 0) sh_cnt = 0;
    __syncthreads();
    // strictly greater first (guaranteed < 64 of them)
    for (int i = tid; i < WIDTH; i += 256) {
        float zv = zrow[i];
        unsigned int key = __float_as_uint(zv > 0.0f ? zv : 0.0f);
        if (key > thresh) {
            unsigned int p = atomicAdd(&sh_cnt, 1u);
            if (p < TOPK) { irow[p] = i; vrow[p] = zv; }
        }
    }
    __syncthreads();
    // then ties at the threshold until 64 filled
    for (int i = tid; i < WIDTH; i += 256) {
        float zv = zrow[i];
        unsigned int key = __float_as_uint(zv > 0.0f ? zv : 0.0f);
        if (key == thresh) {
            unsigned int p = atomicAdd(&sh_cnt, 1u);
            if (p < TOPK) { irow[p] = i; vrow[p] = zv; }
        }
    }
}

// ---------------- decode: out = lam * sum val_i * Adt[idx_i,:] + bd -------
__global__ __launch_bounds__(256)
void sae_decode(const int* __restrict__ idx_in,
                const float* __restrict__ val_in,
                const float* __restrict__ adt,       // (16384, 1024)
                const float* __restrict__ bd,
                const float* __restrict__ lambda_pre,
                float* __restrict__ out) {           // (8192, 1024)
    __shared__ int   sidx[TOPK];
    __shared__ float sval[TOPK];
    const int row = blockIdx.x;
    const int tid = threadIdx.x;                     // 256 -> 4 floats each
    if (tid < TOPK) {
        sidx[tid] = idx_in[row * TOPK + tid];
        sval[tid] = val_in[row * TOPK + tid];
    }
    __syncthreads();

    const float lam = log1pf(expf(lambda_pre[0]));   // softplus

    float a0 = 0.f, a1 = 0.f, a2 = 0.f, a3 = 0.f;
    const float4* adt4 = (const float4*)adt;         // row stride = 256 float4
#pragma unroll 4
    for (int i = 0; i < TOPK; ++i) {
        float v = sval[i];
        float4 w = adt4[sidx[i] * (DIN / 4) + tid];
        a0 += v * w.x; a1 += v * w.y; a2 += v * w.z; a3 += v * w.w;
    }
    float4 b = ((const float4*)bd)[tid];
    float4 o;
    o.x = b.x + lam * a0; o.y = b.y + lam * a1;
    o.z = b.z + lam * a2; o.w = b.w + lam * a3;
    ((float4*)out)[(size_t)row * (DIN / 4) + tid] = o;
}

// ---------------------------------------------------------------------------
extern "C" void kernel_launch(void* const* d_in, const int* in_sizes, int n_in,
                              void* d_out, int out_size, void* d_ws, size_t ws_size,
                              hipStream_t stream) {
    const float* x          = (const float*)d_in[0];   // (8192, 1024)
    const float* Ae         = (const float*)d_in[1];   // (16384, 1024)
    const float* Ad         = (const float*)d_in[2];   // (1024, 16384)
    const float* bd         = (const float*)d_in[3];   // (1, 1024)
    const float* lambda_pre = (const float*)d_in[4];   // (1,)
    float* out = (float*)d_out;

    char* ws = (char*)d_ws;
    unsigned short* xcb = (unsigned short*)(ws + XCB_OFF);
    unsigned short* aeb = (unsigned short*)(ws + AEB_OFF);
    float*          adt = (float*)(ws + ADT_OFF);
    float*          z   = (float*)(ws + Z_OFF);
    int*            tki = (int*)(ws + IDX_OFF);
    float*          tkv = (float*)(ws + VAL_OFF);

    // 1) precision / layout prep
    sae_prep_x <<<(B_ROWS * DIN) / 256, 256, 0, stream>>>(x, bd, xcb);
    sae_prep_ae<<<(WIDTH * DIN) / 256, 256, 0, stream>>>(Ae, aeb);
    sae_trans_ad<<<dim3(WIDTH / 32, DIN / 32), dim3(32, 8), 0, stream>>>(Ad, adt);

    // 2) encode GEMM via bf16 WMMA, async-to-LDS double buffering
    sae_encode<<<dim3(WIDTH / 128, B_ROWS / 128), 256, 0, stream>>>(xcb, aeb, z);

    // 3) per-row top-64 selection (row staged in LDS)
    sae_topk<<<B_ROWS, 256, 0, stream>>>(z, tki, tkv);

    // 4) sparse decode + bias + lambda
    sae_decode<<<B_ROWS, 256, 0, stream>>>(tki, tkv, adt, bd, lambda_pre, out);
}
